// MultiHeadAttention_30296699306172
// MI455X (gfx1250) — compile-verified, matrix operand order checked by reading, not run on hardware
//
#include <hip/hip_runtime.h>
#include <hip/hip_bf16.h>
#include <math.h>

// ---------------------------------------------------------------------------
// Problem constants (match reference)
// ---------------------------------------------------------------------------
#define BATCH     2
#define SEQ       2048
#define DMODEL    1024
#define NHEADS    16
#define HEADDIM   64
#define LN_EPS    1e-5f

#define DEV __device__ __forceinline__

typedef __bf16 bf16_t;
typedef __attribute__((ext_vector_type(16))) __bf16 v16bf;
typedef __attribute__((ext_vector_type(8)))  float  v8f;

// ---------------------------------------------------------------------------
// gfx1250 async global->LDS copies (ASYNCcnt path, CDNA5 ISA).
// lds_off is the 32-bit LDS byte address (low 32 bits of the flat address).
// ---------------------------------------------------------------------------
DEV void async_copy_b128(unsigned lds_off, const void* gaddr) {
    asm volatile("global_load_async_to_lds_b128 %0, %1, off"
                 :: "v"(lds_off), "v"((unsigned long long)(size_t)gaddr)
                 : "memory");
}
DEV void async_copy_b64(unsigned lds_off, const void* gaddr) {
    asm volatile("global_load_async_to_lds_b64 %0, %1, off"
                 :: "v"(lds_off), "v"((unsigned long long)(size_t)gaddr)
                 : "memory");
}
DEV void wait_async0() {
    asm volatile("s_wait_asynccnt 0x0" ::: "memory");
}
DEV unsigned lds_addr(const void* p) {          // flat->LDS: addr[31:0]
    return (unsigned)(size_t)p;
}

// ---------------------------------------------------------------------------
// WMMA fragment loaders, wave32 CDNA5 layouts (ISA 7.12.2):
//   A (16x32 bf16): lane L -> row m = L&15 ; element e -> K via kmap()
//   B (32x16 bf16): lane L -> col n = L&15 ; same K striping
//   C/D (16x16 f32): lane L -> col n = L&15 ; vgpr r -> row m = r + 8*(L>>4)
// ---------------------------------------------------------------------------
DEV int kmap(int lane, int e) {
    return ((e >> 3) << 4) | ((lane >> 4) << 3) | (e & 7);
}

DEV v16bf frag_a_bf16(const bf16_t* __restrict__ base, int ld, int lane, int k0) {
    const int m = lane & 15;
    v16bf a;
#pragma unroll
    for (int e = 0; e < 16; ++e)
        a[e] = base[(size_t)m * ld + k0 + kmap(lane, e)];
    return a;
}

DEV v16bf frag_a_f32(const float* __restrict__ base, int ld, int lane, int k0) {
    const int m = lane & 15;
    v16bf a;
#pragma unroll
    for (int e = 0; e < 16; ++e)
        a[e] = (bf16_t)base[(size_t)m * ld + k0 + kmap(lane, e)];
    return a;
}

// B where memory is indexed [n][k]  (multiply-by-transpose:  X@W^T, Q@K^T)
DEV v16bf frag_b_nk_f32(const float* __restrict__ base, int ld, int lane, int k0) {
    const int n = lane & 15;
    v16bf b;
#pragma unroll
    for (int e = 0; e < 16; ++e)
        b[e] = (bf16_t)base[(size_t)n * ld + k0 + kmap(lane, e)];
    return b;
}
DEV v16bf frag_b_nk_bf16(const bf16_t* __restrict__ base, int ld, int lane, int k0) {
    const int n = lane & 15;
    v16bf b;
#pragma unroll
    for (int e = 0; e < 16; ++e)
        b[e] = base[(size_t)n * ld + k0 + kmap(lane, e)];
    return b;
}
// B where memory is indexed [k][n]  (P@V)
DEV v16bf frag_b_kn_bf16(const bf16_t* __restrict__ base, int ld, int lane, int k0) {
    const int n = lane & 15;
    v16bf b;
#pragma unroll
    for (int e = 0; e < 16; ++e)
        b[e] = base[(size_t)(k0 + kmap(lane, e)) * ld + n];
    return b;
}

DEV v8f wmma_bf16(v16bf a, v16bf b, v8f c) {
    return __builtin_amdgcn_wmma_f32_16x16x32_bf16(
        false, a, false, b, (short)0, c, false, false);
}

// ---------------------------------------------------------------------------
// Kernel 1: projection  Yh[b,h,s,dh] = bf16( X[b,s,:] @ W^T + bias )
// Workgroup = 16 rows x 1024 cols; 8 waves x 8 N-tiles.
// X row-block async-staged to LDS (64 KB), converted once to bf16 (32 KB).
// ---------------------------------------------------------------------------
#define PROJ_LDS_BYTES (16 * DMODEL * 4 + 16 * DMODEL * 2)

__global__ void proj_kernel(const float* __restrict__ x,
                            const float* __restrict__ w,
                            const float* __restrict__ bias,
                            bf16_t* __restrict__ yh) {
    extern __shared__ char psm[];
    float*  xt  = (float*)psm;                       // 16 x 1024 fp32
    bf16_t* xtb = (bf16_t*)(psm + 16 * DMODEL * 4);  // 16 x 1024 bf16

    const int tid  = threadIdx.x;
    const int lane = tid & 31;
    const int wv   = tid >> 5;
    const int row0 = blockIdx.x * 16;                // flat row in [0, B*S)

    // rows are contiguous: one 64 KB async copy, 16 x b128 per thread
    {
        const char* gsrc = (const char*)(x + (size_t)row0 * DMODEL);
        const unsigned l0 = lds_addr(xt);
#pragma unroll
        for (int it = 0; it < 16; ++it) {
            const int j = it * 256 + tid;            // float4 index
            async_copy_b128(l0 + j * 16, gsrc + j * 16);
        }
    }
    wait_async0();
    __syncthreads();
    for (int i = tid; i < 16 * DMODEL; i += 256) xtb[i] = (bf16_t)xt[i];
    __syncthreads();

#pragma unroll
    for (int t = 0; t < 8; ++t) {
        const int n0 = (wv * 8 + t) * 16;
        v8f acc = {};
        for (int k0 = 0; k0 < DMODEL; k0 += 32) {
            __builtin_prefetch(w + (size_t)n0 * DMODEL + k0 + 64, 0, 1);
            v16bf a = frag_a_bf16(xtb, DMODEL, lane, k0);
            v16bf b = frag_b_nk_f32(w + (size_t)n0 * DMODEL, DMODEL, lane, k0);
            acc = wmma_bf16(a, b, acc);
        }
        const int n    = lane & 15;
        const int ncol = n0 + n;
        const int h    = ncol >> 6;
        const int dh   = ncol & (HEADDIM - 1);
        const float bb = bias[ncol];
#pragma unroll
        for (int r = 0; r < 8; ++r) {
            const int m    = r + ((lane >> 4) << 3);
            const int srow = row0 + m;
            const int bi   = srow >> 11;
            const int s    = srow & (SEQ - 1);
            yh[(((size_t)bi * NHEADS + h) * SEQ + s) * HEADDIM + dh] =
                (bf16_t)(acc[r] + bb);
        }
    }
}

// ---------------------------------------------------------------------------
// Kernel 2: scores + mask + softmax + attention write + context
// Workgroup = (b, h, 16 query rows); 8 waves.
// LDS: 128 KB fp32 score stripe + bf16 Q tile + reduction + ctx partials.
// ---------------------------------------------------------------------------
#define ATTN_SC_BYTES   (16 * SEQ * 4)
#define ATTN_QT_BYTES   (16 * HEADDIM * 2)
#define ATTN_RED_BYTES  (16 * 16 * 4)
#define ATTN_CP_BYTES   (2 * 16 * HEADDIM * 4)
#define ATTN_LDS_BYTES  (ATTN_SC_BYTES + ATTN_QT_BYTES + ATTN_RED_BYTES + ATTN_CP_BYTES)

__global__ void attn_kernel(const bf16_t* __restrict__ qh,
                            const bf16_t* __restrict__ kh,
                            const bf16_t* __restrict__ vh,
                            const int*    __restrict__ mask,
                            float* __restrict__ att,
                            float* __restrict__ ctx) {
    extern __shared__ char asm_[];
    float*  sc    = (float*)asm_;                                   // 16 x 2048
    bf16_t* qt    = (bf16_t*)(asm_ + ATTN_SC_BYTES);                // 16 x 64
    float*  red   = (float*)(asm_ + ATTN_SC_BYTES + ATTN_QT_BYTES); // 16 x 16
    float*  cpart = (float*)(asm_ + ATTN_SC_BYTES + ATTN_QT_BYTES + ATTN_RED_BYTES);

    const int tid  = threadIdx.x;
    const int lane = tid & 31;
    const int wv   = tid >> 5;
    const int q0   = blockIdx.x * 16;
    const int h    = blockIdx.y;
    const int b    = blockIdx.z;

    const bf16_t* qbase = qh + (((size_t)b * NHEADS + h) * SEQ + q0) * HEADDIM;
    const bf16_t* kbase = kh + ((size_t)b * NHEADS + h) * SEQ * HEADDIM;
    const bf16_t* vbase = vh + ((size_t)b * NHEADS + h) * SEQ * HEADDIM;

    // ---- async-stage Q tile (16x64 bf16 = 2 KB contiguous) ----
    async_copy_b64(lds_addr(qt) + tid * 8, (const char*)qbase + tid * 8);
    wait_async0();
    __syncthreads();

    // ---- scores = (Q K^T)/8 with mask, into LDS ----
    for (int jt = wv; jt < SEQ / 16; jt += 8) {
        v8f acc = {};
#pragma unroll
        for (int k0 = 0; k0 < HEADDIM; k0 += 32) {
            v16bf a  = frag_a_bf16(qt, HEADDIM, lane, k0);
            v16bf bm = frag_b_nk_bf16(kbase + (size_t)jt * 16 * HEADDIM,
                                      HEADDIM, lane, k0);
            acc = wmma_bf16(a, bm, acc);
        }
        const int n   = lane & 15;
        const int col = jt * 16 + n;
#pragma unroll
        for (int r = 0; r < 8; ++r) {
            const int m  = r + ((lane >> 4) << 3);
            const int mv = mask[((size_t)b * SEQ + (q0 + m)) * SEQ + col];
            const float v = acc[r] * 0.125f;      // 1/sqrt(HEADDIM)
            sc[(size_t)m * SEQ + col] = (mv == 0) ? -1.0e9f : v;
        }
    }
    __syncthreads();

    // ---- softmax over LDS rows; single streaming write of probabilities ----
    {
        const int row = tid >> 4;                 // 16 threads per row
        const int seg = tid & 15;                 // 128 cols each
        float* srow = sc + (size_t)row * SEQ;
        const int c0 = seg * 128;

        float mx = -3.4e38f;
        for (int c = 0; c < 128; ++c) mx = fmaxf(mx, srow[c0 + c]);
        red[row * 16 + seg] = mx;
        __syncthreads();
        float rmx = -3.4e38f;
#pragma unroll
        for (int i = 0; i < 16; ++i) rmx = fmaxf(rmx, red[row * 16 + i]);
        __syncthreads();

        float sum = 0.0f;
        for (int c = 0; c < 128; ++c) {
            const float e = expf(srow[c0 + c] - rmx);
            srow[c0 + c] = e;
            sum += e;
        }
        red[row * 16 + seg] = sum;
        __syncthreads();
        float rs = 0.0f;
#pragma unroll
        for (int i = 0; i < 16; ++i) rs += red[row * 16 + i];
        const float inv = 1.0f / rs;

        float* arow = att + (((size_t)b * NHEADS + h) * SEQ + (q0 + row)) * SEQ;
        for (int c = 0; c < 128; ++c) {
            const float p = srow[c0 + c] * inv;
            srow[c0 + c] = p;                     // keep probs for P@V
            arow[c0 + c] = p;
        }
    }
    __syncthreads();

    // ---- context = P @ V : all 8 waves, K=2048 split into two halves ----
    {
        const int d0    = (wv & 3) * 16;
        const int khalf = wv >> 2;
        v8f acc = {};
        for (int k0 = khalf * (SEQ / 2); k0 < (khalf + 1) * (SEQ / 2); k0 += 32) {
            v16bf a  = frag_a_f32(sc, SEQ, lane, k0);
            v16bf bm = frag_b_kn_bf16(vbase + d0, HEADDIM, lane, k0);
            acc = wmma_bf16(a, bm, acc);
        }
        const int n = lane & 15;
#pragma unroll
        for (int r = 0; r < 8; ++r) {
            const int m = r + ((lane >> 4) << 3);
            cpart[((size_t)(khalf * 16 + m)) * HEADDIM + d0 + n] = acc[r];
        }
    }
    __syncthreads();
    if (wv < 4) {
        const int d0 = wv * 16;
        const int n  = lane & 15;
#pragma unroll
        for (int r = 0; r < 8; ++r) {
            const int m = r + ((lane >> 4) << 3);
            const float s = cpart[(size_t)m * HEADDIM + d0 + n] +
                            cpart[(size_t)(16 + m) * HEADDIM + d0 + n];
            ctx[((size_t)b * SEQ + q0 + m) * DMODEL + h * HEADDIM + d0 + n] = s;
        }
    }
}

// ---------------------------------------------------------------------------
// Kernel 3: out = LN(residual + ctx @ Wo^T + bo)
// ctx async-staged fp32 (64 KB) -> bf16 (32 KB); pre-LN buffer aliases the
// dead fp32 staging region.  Total LDS ~97 KB.
// ---------------------------------------------------------------------------
#define LN_LDS_BYTES (16 * DMODEL * 4 + 16 * DMODEL * 2 + 16 * 16 * 4)

__global__ void outproj_ln_kernel(const float* __restrict__ ctx,
                                  const float* __restrict__ wo,
                                  const float* __restrict__ bo,
                                  const float* __restrict__ resid,
                                  const float* __restrict__ gamma,
                                  const float* __restrict__ beta,
                                  float* __restrict__ out) {
    extern __shared__ char lsm[];
    float*  xt  = (float*)lsm;                       // 16 x 1024 fp32 stage
    float*  yb  = (float*)lsm;                       // aliases xt (dead after cvt)
    bf16_t* xtb = (bf16_t*)(lsm + 16 * DMODEL * 4);  // 16 x 1024 bf16
    float*  red = (float*)(lsm + 16 * DMODEL * 4 + 16 * DMODEL * 2); // 16 x 16

    const int tid  = threadIdx.x;
    const int lane = tid & 31;
    const int wv   = tid >> 5;
    const int row0 = blockIdx.x * 16;

    {
        const char* gsrc = (const char*)(ctx + (size_t)row0 * DMODEL);
        const unsigned l0 = lds_addr(xt);
#pragma unroll
        for (int it = 0; it < 16; ++it) {
            const int j = it * 256 + tid;
            async_copy_b128(l0 + j * 16, gsrc + j * 16);
        }
    }
    wait_async0();
    __syncthreads();
    for (int i = tid; i < 16 * DMODEL; i += 256) xtb[i] = (bf16_t)xt[i];
    __syncthreads();                                 // xt dead; yb may be written

#pragma unroll
    for (int t = 0; t < 8; ++t) {
        const int n0 = (wv * 8 + t) * 16;
        v8f acc = {};
        for (int k0 = 0; k0 < DMODEL; k0 += 32) {
            __builtin_prefetch(wo + (size_t)n0 * DMODEL + k0 + 64, 0, 1);
            v16bf a = frag_a_bf16(xtb, DMODEL, lane, k0);
            v16bf b = frag_b_nk_f32(wo + (size_t)n0 * DMODEL, DMODEL, lane, k0);
            acc = wmma_bf16(a, b, acc);
        }
        const int n    = lane & 15;
        const int ncol = n0 + n;
        const float bb = bo[ncol];
#pragma unroll
        for (int r = 0; r < 8; ++r) {
            const int m = r + ((lane >> 4) << 3);
            yb[(size_t)m * DMODEL + ncol] =
                acc[r] + bb + resid[(size_t)(row0 + m) * DMODEL + ncol];
        }
    }
    __syncthreads();

    // ---- fused LayerNorm over each 1024-wide row ----
    {
        const int row = tid >> 4;
        const int seg = tid & 15;
        const float* yrow = yb + (size_t)row * DMODEL;
        const int c0 = seg * 64;

        float ps = 0.0f;
        for (int c = 0; c < 64; ++c) ps += yrow[c0 + c];
        red[row * 16 + seg] = ps;
        __syncthreads();
        float mu = 0.0f;
#pragma unroll
        for (int i = 0; i < 16; ++i) mu += red[row * 16 + i];
        mu *= (1.0f / DMODEL);
        __syncthreads();

        float pv = 0.0f;
        for (int c = 0; c < 64; ++c) {
            const float d = yrow[c0 + c] - mu;
            pv += d * d;
        }
        red[row * 16 + seg] = pv;
        __syncthreads();
        float var = 0.0f;
#pragma unroll
        for (int i = 0; i < 16; ++i) var += red[row * 16 + i];
        var *= (1.0f / DMODEL);
        const float rstd = rsqrtf(var + LN_EPS);

        float* orow = out + (size_t)(row0 + row) * DMODEL;
        for (int c = 0; c < 64; ++c) {
            const int col = c0 + c;
            orow[col] = (yrow[col] - mu) * rstd * gamma[col] + beta[col];
        }
    }
}

// ---------------------------------------------------------------------------
// Host launcher
// ---------------------------------------------------------------------------
extern "C" void kernel_launch(void* const* d_in, const int* in_sizes, int n_in,
                              void* d_out, int out_size, void* d_ws, size_t ws_size,
                              hipStream_t stream) {
    (void)in_sizes; (void)n_in; (void)out_size; (void)ws_size;

    const float* q    = (const float*)d_in[0];
    const float* k    = (const float*)d_in[1];
    const float* v    = (const float*)d_in[2];
    const int*   msk  = (const int*)d_in[3];
    const float* wq   = (const float*)d_in[4];
    const float* bq   = (const float*)d_in[5];
    const float* wk   = (const float*)d_in[6];
    const float* bk   = (const float*)d_in[7];
    const float* wv_  = (const float*)d_in[8];
    const float* bv   = (const float*)d_in[9];
    const float* wo   = (const float*)d_in[10];
    const float* bo   = (const float*)d_in[11];
    const float* gmm  = (const float*)d_in[12];
    const float* bet  = (const float*)d_in[13];

    const size_t BSD = (size_t)BATCH * SEQ * DMODEL;   // 4,194,304 elems

    float* out = (float*)d_out;                        // [B,S,D] fp32
    float* att = out + BSD;                            // [B,H,S,S] fp32

    bf16_t* qh  = (bf16_t*)d_ws;                       // [B,H,S,dh] bf16
    bf16_t* kh  = qh + BSD;
    bf16_t* vh  = kh + BSD;
    float*  ctx = (float*)(vh + BSD);                  // [B,S,D] fp32

    const dim3 blk(256);
    const int  rowBlocks = (BATCH * SEQ) / 16;         // 256

    proj_kernel<<<dim3(rowBlocks), blk, PROJ_LDS_BYTES, stream>>>(q, wq, bq, qh);
    proj_kernel<<<dim3(rowBlocks), blk, PROJ_LDS_BYTES, stream>>>(k, wk, bk, kh);
    proj_kernel<<<dim3(rowBlocks), blk, PROJ_LDS_BYTES, stream>>>(v, wv_, bv, vh);

    attn_kernel<<<dim3(SEQ / 16, NHEADS, BATCH), blk, ATTN_LDS_BYTES, stream>>>(
        qh, kh, vh, msk, att, ctx);

    outproj_ln_kernel<<<dim3(rowBlocks), blk, LN_LDS_BYTES, stream>>>(
        ctx, wo, bo, q, gmm, bet, out);
}
